// GNNModel_68934225101284
// MI455X (gfx1250) — matrix operand order repaired and checked by reading, not compiled, not run on hardware
//
#include <hip/hip_runtime.h>
#include <hip/hip_bf16.h>

#define HD 128
#define NEGS 0.01f
#define EPS 1e-5f

typedef __attribute__((ext_vector_type(16))) _Float16 v16h;
typedef __attribute__((ext_vector_type(8)))  float    v8f;

// ---------------------------------------------------------------- helpers
__device__ inline float lrelu(float v) { return v >= 0.0f ? v : NEGS * v; }

// order-preserving float <-> uint map for atomic max on floats
__device__ inline unsigned f2mono(float f) {
  unsigned u = __float_as_uint(f);
  return (u & 0x80000000u) ? ~u : (u | 0x80000000u);
}
__device__ inline float mono2f(unsigned u) {
  return (u & 0x80000000u) ? __uint_as_float(u ^ 0x80000000u) : __uint_as_float(~u);
}
#define MONO_NEG_INF 0x007FFFFFu   // f2mono(-inf): identity for max

__device__ inline v8f wmma16x16x32(v16h a, v16h b, v8f c) {
  return __builtin_amdgcn_wmma_f32_16x16x32_f16(
      /*neg_a=*/false, a, /*neg_b=*/false, b,
      /*c_mod=*/(short)0, c, /*reuse_a=*/false, /*reuse_b=*/false);
}

// A fragment (16x32 f16) from LDS row-major [16][ldk] halves, K-offset k0.
// CDNA5 ISA 7.12.2 16-bit A layout: lanes 0-15 hold K 0-7/16-23, lanes 16-31 K 8-15/24-31.
// Per lane the two 8-half runs are contiguous -> 2x ds_load_b128.
__device__ inline v16h frag_a_lds(const _Float16* s, int ldk, int k0, int lane) {
  const int g = (lane >> 4) & 1;
  const int m = lane & 15;
  v16h a;
#pragma unroll
  for (int p = 0; p < 16; ++p) {
    const int v = p >> 1;
    const int k = ((v < 4) ? 0 : 16) + ((v & 3) << 1) + (p & 1) + (g << 3);
    a[p] = s[m * ldk + k0 + k];
  }
  return a;
}

// B fragment from fragment-major packed weights: fragment f = (t * ksteps + s),
// lane's 16 halves contiguous at ((f*32 + lane) * 16) -> 2x global_load_b128, coalesced.
__device__ inline v16h frag_b_pk(const _Float16* wf, int ksteps, int t, int s, int lane) {
  return *(const v16h*)(wf + (size_t)(((t * ksteps) + s) * 32 + lane) * 16);
}

// ---------------------------------------------------------------- utility kernels
__global__ void k_fill_u32(unsigned* p, unsigned v, int n) {
  int i = blockIdx.x * blockDim.x + threadIdx.x;
  if (i < n) p[i] = v;
}

// pack f32 row-major [K][128] weights into WMMA fragment-major f16 layout.
// half index i = ((t*ksteps + s)*32 + lane)*16 + p ;
// source element: k = 32*s + 16*(p>>3) + 8*(lane>>4) + (p&7) , col = 16*t + (lane&15)
__global__ void k_pack_b(const float* w, _Float16* wf, int K) {
  int i = blockIdx.x * blockDim.x + threadIdx.x;
  if (i < K * HD) {
    int p = i & 15;
    int lane = (i >> 4) & 31;
    int f = i >> 9;
    int ksteps = K >> 5;
    int t = f / ksteps, s = f - t * ksteps;
    int k = (s << 5) + ((p >> 3) << 4) + ((lane >> 4) << 3) + (p & 7);
    int col = (t << 4) + (lane & 15);
    wf[i] = (_Float16)w[k * HD + col];
  }
}

__global__ void k_ranges_init(int* sg, int* eg, int G) {
  int g = blockIdx.x * blockDim.x + threadIdx.x;
  if (g < G) { sg[g] = 0x7FFFFFFF; eg[g] = 0; }
}

__global__ void k_ranges_scatter(const int* batch, int* sg, int* eg, int N) {
  int i = blockIdx.x * blockDim.x + threadIdx.x;
  if (i < N) {
    int g = batch[i];
    atomicMin(&sg[g], i);
    atomicMax(&eg[g], i + 1);
  }
}

__global__ void k_deg(const int* dst, float* deg, int E) {
  int i = blockIdx.x * blockDim.x + threadIdx.x;
  if (i < E) atomicAdd(&deg[dst[i]], 1.0f);
}

// input_net first linear: C_IN=2 -> 128
__global__ void k_input_linear(const float* x, const float* w, const float* b,
                               float* h, int N) {
  int i = blockIdx.x * blockDim.x + threadIdx.x;
  if (i < N * HD) {
    int n = i >> 7, f = i & (HD - 1);
    h[i] = x[n * 2 + 0] * w[0 * HD + f] + x[n * 2 + 1] * w[1 * HD + f] + b[f];
  }
}

// ---------------------------------------------------------------- GraphNorm
__global__ void k_gn_mean(const float* x, const int* sg, const int* eg, float* mean) {
  int g = blockIdx.x, f = threadIdx.x;
  int s = sg[g], e = eg[g];
  float sum = 0.0f;
  for (int i = s; i < e; ++i) sum += x[i * HD + f];
  float c = (float)(e - s);
  if (c < 1.0f) c = 1.0f;
  mean[g * HD + f] = sum / c;
}

__global__ void k_gn_var(const float* x, const int* sg, const int* eg,
                         const float* mean, const float* ms, float* istd) {
  int g = blockIdx.x, f = threadIdx.x;
  int s = sg[g], e = eg[g];
  float mu = mean[g * HD + f] * ms[f];
  float ss = 0.0f;
  for (int i = s; i < e; ++i) {
    float o = x[i * HD + f] - mu;
    ss += o * o;
  }
  float c = (float)(e - s);
  if (c < 1.0f) c = 1.0f;
  istd[g * HD + f] = rsqrtf(ss / c + EPS);
}

__global__ void k_gn_apply(const float* x, const int* batch, const float* mean,
                           const float* ms, const float* istd, const float* w,
                           const float* b, float* out, int N, int act) {
  int i = blockIdx.x * blockDim.x + threadIdx.x;
  if (i < N * HD) {
    int n = i >> 7, f = i & (HD - 1);
    int g = batch[n];
    float o = (x[i] - mean[g * HD + f] * ms[f]) * istd[g * HD + f] * w[f] + b[f];
    out[i] = act ? lrelu(o) : o;
  }
}

// ---------------------------------------------------------------- WMMA node GEMM
// out[N,128] = A[N,K] @ W[K,128] + bias (+resid)
// MODE 0: A = srcA ; MODE 1: A = [decode(max), sum/deg], residual add
template <int K, int MODE>
__global__ void k_gemm_nodes(const float* srcA, const unsigned* amax, const float* asum,
                             const float* deg, const _Float16* wf, const float* bias,
                             const float* resid, float* out, int N) {
  __shared__ _Float16 sA[16 * K];
  const int row0 = blockIdx.x * 16;
  const int tid = threadIdx.x;
  const int lane = tid & 31, wave = tid >> 5;
  constexpr int KSTEPS = K >> 5;
  constexpr int KSH = (K == 256) ? 8 : 7;

#pragma unroll
  for (int it = 0; it < (16 * K) / 256; ++it) {
    int idx = it * 256 + tid;
    int m = idx >> KSH, k = idx & (K - 1);
    int row = row0 + m;
    float v = 0.0f;
    if (row < N) {
      if (MODE == 0) {
        v = srcA[row * K + k];
      } else {
        if (k < HD) {
          unsigned u = amax[row * HD + k];
          v = (u == MONO_NEG_INF) ? 0.0f : mono2f(u);  // empty segment -> 0
        } else {
          float d = deg[row];
          if (d < 1.0f) d = 1.0f;
          v = asum[row * HD + (k - HD)] / d;
        }
      }
    }
    sA[m * K + k] = (_Float16)v;
  }
  __syncthreads();

  v8f acc = {};
#pragma unroll
  for (int s = 0; s < KSTEPS; ++s) {
    v16h a = frag_a_lds(sA, K, s << 5, lane);
    v16h b = frag_b_pk(wf, KSTEPS, wave, s, lane);
    acc = wmma16x16x32(a, b, acc);
  }

  const int col = wave * 16 + (lane & 15), g = (lane >> 4) & 1;
  const float bb = bias[col];
#pragma unroll
  for (int r = 0; r < 8; ++r) {
    int row = row0 + r + 8 * g;
    if (row < N) {
      float v = acc[r] + bb;
      if (MODE == 1) v += resid[row * HD + col];
      out[row * HD + col] = v;
    }
  }
}

// ---------------------------------------------------------------- fused EdgeConv
// e = W2 @ lrelu(W1 @ [x_i, x_j - x_i] + b1) + b2 ; scatter max/sum into dst
__global__ void k_edgeconv(const float* hn, const int* src, const int* dst,
                           const _Float16* w1f, const float* b1,
                           const _Float16* w2f, const float* b2,
                           unsigned* amax, float* asum, int E) {
  __shared__ _Float16 sA[16 * 256];  // [x_i, x_j - x_i] as f16
  __shared__ _Float16 sE[16 * 128];  // hidden activations as f16
  __shared__ int sSrc[16], sDst[16];

  const int e0 = blockIdx.x * 16;
  const int tid = threadIdx.x;
  const int lane = tid & 31, wave = tid >> 5;

  if (tid < 16) {
    int e = e0 + tid;
    sSrc[tid] = (e < E) ? src[e] : -1;
    sDst[tid] = (e < E) ? dst[e] : -1;
    // prefetch next block's gather rows into cache (global_prefetch path)
    int en = e + 16;
    if (en < E) {
      __builtin_prefetch(&hn[(size_t)src[en] * HD], 0, 1);
      __builtin_prefetch(&hn[(size_t)dst[en] * HD], 0, 1);
    }
  }
  __syncthreads();

#pragma unroll
  for (int it = 0; it < 16; ++it) {
    int idx = it * 256 + tid;
    int m = idx >> 8, c = idx & 255;
    int di = sDst[m], si = sSrc[m];
    float v = 0.0f;
    if (di >= 0) {
      float xi = hn[di * HD + (c & (HD - 1))];
      v = (c < HD) ? xi : (hn[si * HD + (c & (HD - 1))] - xi);
    }
    sA[idx] = (_Float16)v;
  }
  __syncthreads();

  const int col = wave * 16 + (lane & 15), g = (lane >> 4) & 1;

  // GEMM1: [16,256] @ [256,128] -> lrelu -> LDS (f16)
  v8f acc = {};
#pragma unroll
  for (int s = 0; s < 8; ++s) {
    v16h a = frag_a_lds(sA, 256, s << 5, lane);
    v16h b = frag_b_pk(w1f, 8, wave, s, lane);
    acc = wmma16x16x32(a, b, acc);
  }
  {
    float bb = b1[col];
#pragma unroll
    for (int r = 0; r < 8; ++r) {
      sE[(r + 8 * g) * HD + col] = (_Float16)lrelu(acc[r] + bb);
    }
  }
  __syncthreads();

  // GEMM2: [16,128] @ [128,128]
  v8f acc2 = {};
#pragma unroll
  for (int s = 0; s < 4; ++s) {
    v16h a = frag_a_lds(sE, HD, s << 5, lane);
    v16h b = frag_b_pk(w2f, 4, wave, s, lane);
    acc2 = wmma16x16x32(a, b, acc2);
  }
  {
    float bb = b2[col];
#pragma unroll
    for (int r = 0; r < 8; ++r) {
      int di = sDst[r + 8 * g];
      if (di >= 0) {
        float v = acc2[r] + bb;
        atomicMax(&amax[di * HD + col], f2mono(v));
        atomicAdd(&asum[di * HD + col], v);
      }
    }
  }
}

// ---------------------------------------------------------------- final GN + pooling
__global__ void k_pool(const float* h, const int* sg, const int* eg,
                       const float* mean, const float* ms, const float* istd,
                       const float* w, const float* b, float* z) {
  int g = blockIdx.x, f = threadIdx.x;
  int s = sg[g], e = eg[g];
  float mu = mean[g * HD + f] * ms[f], is = istd[g * HD + f];
  float ww = w[f], bb = b[f];
  float mx = -3.402823466e38f, sum = 0.0f;
  for (int i = s; i < e; ++i) {
    float y = lrelu((h[i * HD + f] - mu) * is * ww + bb);
    mx = fmaxf(mx, y);
    sum += y;
  }
  float c = (float)(e - s);
  if (c < 1.0f) c = 1.0f;
  if (e <= s) mx = 0.0f;  // empty graph: segment_max -> 0
  z[g * 256 + f] = mx;
  z[g * 256 + HD + f] = sum / c;
}

// ---------------------------------------------------------------- tiny classifier
__global__ void k_linear(const float* in, const float* w, const float* b,
                         float* out, int M, int K, int Nout) {
  int i = blockIdx.x * blockDim.x + threadIdx.x;
  if (i < M * Nout) {
    int m = i / Nout, n = i % Nout;
    float acc = b[n];
    for (int k = 0; k < K; ++k) acc += in[m * K + k] * w[k * Nout + n];
    out[i] = acc;
  }
}

__global__ void k_ln_lrelu(const float* x, const float* w, const float* b,
                           float* out, int D) {
  extern __shared__ float red[];
  int m = blockIdx.x, f = threadIdx.x;
  float v = x[m * D + f];
  red[f] = v;
  __syncthreads();
  for (int s = D >> 1; s > 0; s >>= 1) {
    if (f < s) red[f] += red[f + s];
    __syncthreads();
  }
  float mu = red[0] / (float)D;
  __syncthreads();
  float d = v - mu;
  red[f] = d * d;
  __syncthreads();
  for (int s = D >> 1; s > 0; s >>= 1) {
    if (f < s) red[f] += red[f + s];
    __syncthreads();
  }
  float var = red[0] / (float)D;
  out[m * D + f] = lrelu(w[f] * d * rsqrtf(var + EPS) + b[f]);
}

// ---------------------------------------------------------------- host launch
extern "C" void kernel_launch(void* const* d_in, const int* in_sizes, int n_in,
                              void* d_out, int out_size, void* d_ws, size_t ws_size,
                              hipStream_t stream) {
  (void)n_in; (void)out_size; (void)ws_size;
  const float* x = (const float*)d_in[0];
  const int N = in_sizes[0] / 2;
  const int* ei = (const int*)d_in[1];
  const int E = in_sizes[1] / 2;
  const int* src = ei;
  const int* dst = ei + E;
  const int* batch = (const int*)d_in[2];
  const int G = 64;  // reference num_graphs

  // params: jax tree order (dict keys sorted alphabetically)
  const float* c1_b   = (const float*)d_in[4];
  const float* c1_w   = (const float*)d_in[5];
  const float* c2_b   = (const float*)d_in[6];
  const float* c2_w   = (const float*)d_in[7];
  const float* c3_b   = (const float*)d_in[8];
  const float* c3_w   = (const float*)d_in[9];
  const float* fgn_b  = (const float*)d_in[10];
  const float* fgn_ms = (const float*)d_in[11];
  const float* fgn_w  = (const float*)d_in[12];
  const float* ign_b  = (const float*)d_in[13];
  const float* ign_ms = (const float*)d_in[14];
  const float* ign_w  = (const float*)d_in[15];
  const float* il1_b  = (const float*)d_in[16];
  const float* il1_w  = (const float*)d_in[17];
  const float* il2_b  = (const float*)d_in[18];
  const float* il2_w  = (const float*)d_in[19];
  struct Lp { const float *w1b,*w1,*w2b,*w2,*gnb,*gnms,*gnw,*pb,*pw; } L[3];
  for (int i = 0; i < 3; ++i) {
    const int o = 20 + 9 * i;
    L[i].w1b = (const float*)d_in[o + 0];
    L[i].w1  = (const float*)d_in[o + 1];
    L[i].w2b = (const float*)d_in[o + 2];
    L[i].w2  = (const float*)d_in[o + 3];
    L[i].gnb = (const float*)d_in[o + 4];
    L[i].gnms= (const float*)d_in[o + 5];
    L[i].gnw = (const float*)d_in[o + 6];
    L[i].pb  = (const float*)d_in[o + 7];
    L[i].pw  = (const float*)d_in[o + 8];
  }
  const float* ln1_b = (const float*)d_in[47];
  const float* ln1_w = (const float*)d_in[48];
  const float* ln2_b = (const float*)d_in[49];
  const float* ln2_w = (const float*)d_in[50];

  // workspace carving (256B aligned)
  char* wp = (char*)d_ws;
  auto carve = [&](size_t bytes) -> void* {
    void* r = (void*)wp;
    wp += (bytes + 255) & ~(size_t)255;
    return r;
  };
  float*    h    = (float*)carve(sizeof(float) * (size_t)N * HD);
  float*    hn   = (float*)carve(sizeof(float) * (size_t)N * HD);
  unsigned* amx  = (unsigned*)carve(sizeof(unsigned) * (size_t)N * HD);
  float*    asum = (float*)carve(sizeof(float) * (size_t)N * HD);
  float*    deg  = (float*)carve(sizeof(float) * (size_t)N);
  int*      sg   = (int*)carve(sizeof(int) * G);
  int*      eg   = (int*)carve(sizeof(int) * G);
  float*    mean = (float*)carve(sizeof(float) * G * HD);
  float*    istd = (float*)carve(sizeof(float) * G * HD);
  float*    z    = (float*)carve(sizeof(float) * G * 256);
  float*    t1   = (float*)carve(sizeof(float) * G * HD);
  float*    t2   = (float*)carve(sizeof(float) * G * 64);
  _Float16* hil2 = (_Float16*)carve(sizeof(_Float16) * 128 * 128);
  _Float16 *hw1[3], *hw2[3], *hwp[3];
  for (int i = 0; i < 3; ++i) {
    hw1[i] = (_Float16*)carve(sizeof(_Float16) * 256 * 128);
    hw2[i] = (_Float16*)carve(sizeof(_Float16) * 128 * 128);
    hwp[i] = (_Float16*)carve(sizeof(_Float16) * 256 * 128);
  }

  const int TB = 256;
  // fragment-major f16 weight packing
  k_pack_b<<<(128 * 128 + TB - 1) / TB, TB, 0, stream>>>(il2_w, hil2, 128);
  for (int i = 0; i < 3; ++i) {
    k_pack_b<<<(256 * 128 + TB - 1) / TB, TB, 0, stream>>>(L[i].w1, hw1[i], 256);
    k_pack_b<<<(128 * 128 + TB - 1) / TB, TB, 0, stream>>>(L[i].w2, hw2[i], 128);
    k_pack_b<<<(256 * 128 + TB - 1) / TB, TB, 0, stream>>>(L[i].pw, hwp[i], 256);
  }

  // graph ranges + degrees
  k_ranges_init<<<1, G, 0, stream>>>(sg, eg, G);
  k_ranges_scatter<<<(N + TB - 1) / TB, TB, 0, stream>>>(batch, sg, eg, N);
  k_fill_u32<<<(N + TB - 1) / TB, TB, 0, stream>>>((unsigned*)deg, 0u, N);
  k_deg<<<(E + TB - 1) / TB, TB, 0, stream>>>(dst, deg, E);

  const int NH = N * HD;
  // input net: lin -> gn -> lrelu -> lin
  k_input_linear<<<(NH + TB - 1) / TB, TB, 0, stream>>>(x, il1_w, il1_b, h, N);
  k_gn_mean<<<G, HD, 0, stream>>>(h, sg, eg, mean);
  k_gn_var<<<G, HD, 0, stream>>>(h, sg, eg, mean, ign_ms, istd);
  k_gn_apply<<<(NH + TB - 1) / TB, TB, 0, stream>>>(h, batch, mean, ign_ms, istd,
                                                    ign_w, ign_b, hn, N, 1);
  k_gemm_nodes<128, 0><<<(N + 15) / 16, TB, 0, stream>>>(
      hn, nullptr, nullptr, nullptr, hil2, il2_b, nullptr, h, N);

  // 3 EdgeConv residual blocks
  for (int i = 0; i < 3; ++i) {
    k_gn_mean<<<G, HD, 0, stream>>>(h, sg, eg, mean);
    k_gn_var<<<G, HD, 0, stream>>>(h, sg, eg, mean, L[i].gnms, istd);
    k_gn_apply<<<(NH + TB - 1) / TB, TB, 0, stream>>>(h, batch, mean, L[i].gnms, istd,
                                                      L[i].gnw, L[i].gnb, hn, N, 1);
    k_fill_u32<<<(NH + TB - 1) / TB, TB, 0, stream>>>(amx, MONO_NEG_INF, NH);
    k_fill_u32<<<(NH + TB - 1) / TB, TB, 0, stream>>>((unsigned*)asum, 0u, NH);
    k_edgeconv<<<(E + 15) / 16, TB, 0, stream>>>(hn, src, dst, hw1[i], L[i].w1b,
                                                 hw2[i], L[i].w2b, amx, asum, E);
    k_gemm_nodes<256, 1><<<(N + 15) / 16, TB, 0, stream>>>(
        nullptr, amx, asum, deg, hwp[i], L[i].pb, h, h, N);
  }

  // final gn + lrelu + pooling
  k_gn_mean<<<G, HD, 0, stream>>>(h, sg, eg, mean);
  k_gn_var<<<G, HD, 0, stream>>>(h, sg, eg, mean, fgn_ms, istd);
  k_pool<<<G, HD, 0, stream>>>(h, sg, eg, mean, fgn_ms, istd, fgn_w, fgn_b, z);

  // classifier
  k_linear<<<(G * 128 + TB - 1) / TB, TB, 0, stream>>>(z, c1_w, c1_b, t1, G, 256, 128);
  k_ln_lrelu<<<G, 128, 128 * sizeof(float), stream>>>(t1, ln1_w, ln1_b, t1, 128);
  k_linear<<<(G * 64 + TB - 1) / TB, TB, 0, stream>>>(t1, c2_w, c2_b, t2, G, 128, 64);
  k_ln_lrelu<<<G, 64, 64 * sizeof(float), stream>>>(t2, ln2_w, ln2_b, t2, 64);
  k_linear<<<1, TB, 0, stream>>>(t2, c3_w, c3_b, (float*)d_out, G, 64, 1);
}
